// VQVAEDensity_68478958567988
// MI455X (gfx1250) — matrix operand order repaired and checked by reading, not compiled, not run on hardware
//
#include <hip/hip_runtime.h>
#include <stdint.h>

// ---------------------------------------------------------------------------
// VQ-VAE forward for MI455X (gfx1250, wave32, WMMA + async global->LDS).
// GEMMs: bf16x3 split precision on v_wmma_f32_16x16x32_bf16, block tile
// 128x128x32, 8 waves (2x4), wave tile 64x32. Tiles stream into LDS with
// GLOBAL_LOAD_ASYNC_TO_LDS_B128 (ASYNCcnt), double-buffered so tile k+1 copy
// overlaps tile k WMMAs; one workgroup barrier per K-step.
// VQ distance GEMM fuses argmin via LDS tile + packed u64 atomicMin.
// ---------------------------------------------------------------------------

typedef __bf16 bf16;
typedef __attribute__((ext_vector_type(16))) __bf16 bf16x16;
typedef __attribute__((ext_vector_type(8)))  float   floatx8;
typedef __attribute__((ext_vector_type(4)))  int     i32x4;

#define AS_GLOBAL __attribute__((address_space(1)))
#define AS_LDS    __attribute__((address_space(3)))

static constexpr int BM = 128, BN = 128, BK = 32;
static constexpr int LSTR = 40;          // bf16 elems per LDS row: 32 + 8 pad (80B, 16B aligned)
static constexpr int TB  = 16384;        // batch
static constexpr int DIN = 1024, HH = 2048, DC = 128, KCB = 4096;

struct GemmSmem {
  alignas(16) bf16 ahi[BM][LSTR];
  alignas(16) bf16 alo[BM][LSTR];
  alignas(16) bf16 bhi[BN][LSTR];
  alignas(16) bf16 blo[BN][LSTR];
};                                        // 40960 B

// ---------------- device helpers ----------------

__device__ __forceinline__ unsigned fkey(float f) {
  // monotonic float -> uint mapping (argmin via integer min)
  unsigned u = __float_as_uint(f);
  return u ^ ((unsigned)((int)u >> 31) | 0x80000000u);
}

__device__ __forceinline__ void wait_async0() {
  asm volatile("s_wait_asynccnt 0x0" ::: "memory");
}

// Async-stage a 128x32 bf16 tile from global (row stride ldK elems) into LDS.
// 512 x 16B chunks over 256 threads -> 2 async b128 issues per thread.
__device__ __forceinline__ void stage_tile_async(const bf16* __restrict__ g, int ldK,
                                                 bf16 (*s)[LSTR], int tid) {
#pragma unroll
  for (int p = 0; p < 2; ++p) {
    int c  = tid + p * 256;
    int r  = c >> 2;
    int cc = c & 3;
    __builtin_amdgcn_global_load_async_to_lds_b128(
        (AS_GLOBAL i32x4*)(g + (long)r * ldK + cc * 8),
        (AS_LDS i32x4*)(&s[r][cc * 8]),
        0, 0);
  }
}

__device__ __forceinline__ void stage_all_async(
    const bf16* __restrict__ Ahi, const bf16* __restrict__ Alo,
    const bf16* __restrict__ Bthi, const bf16* __restrict__ Btlo,
    GemmSmem& s, int K, long blockRow, long blockCol, int k0, int tid) {
  stage_tile_async(Ahi  + blockRow * K + k0, K, s.ahi, tid);
  stage_tile_async(Alo  + blockRow * K + k0, K, s.alo, tid);
  stage_tile_async(Bthi + blockCol * K + k0, K, s.bhi, tid);
  stage_tile_async(Btlo + blockCol * K + k0, K, s.blo, tid);
}

// A fragment (16x32 bf16): lanes 0-15 hold K {0..7,16..23}, lanes 16-31 {8..15,24..31}
__device__ __forceinline__ bf16x16 load_fragA(const bf16* rowp, int lane) {
  const int kb = (lane >> 4) * 8;
  union { uint4 u[2]; bf16x16 v; } f;
  f.u[0] = *reinterpret_cast<const uint4*>(rowp + kb);
  f.u[1] = *reinterpret_cast<const uint4*>(rowp + kb + 16);
  return f.v;
}

// B fragment (32x16 bf16, fed as rows of B^T): lanes 0-15 hold K {0..7,16..23}, lanes 16-31 {8..15,24..31}
__device__ __forceinline__ bf16x16 load_fragB(const bf16* rowp, int lane) {
  const int kb = (lane >> 4) * 8;
  union { uint4 u[2]; bf16x16 v; } f;
  f.u[0] = *reinterpret_cast<const uint4*>(rowp + kb);
  f.u[1] = *reinterpret_cast<const uint4*>(rowp + kb + 16);
  return f.v;
}

// bf16x3 double-buffered main loop shared by all GEMM kernels.
__device__ __forceinline__ void gemm_mainloop(
    const bf16* __restrict__ Ahi, const bf16* __restrict__ Alo,
    const bf16* __restrict__ Bthi, const bf16* __restrict__ Btlo,
    GemmSmem* st, floatx8 acc[4][2],
    int tid, int K, long blockRow, long blockCol) {
  const int lane  = tid & 31;
  const int waveM = (tid >> 5) >> 2;
  const int waveN = (tid >> 5) & 3;
  const int nk = K / BK;

  stage_all_async(Ahi, Alo, Bthi, Btlo, st[0], K, blockRow, blockCol, 0, tid);

  for (int kt = 0; kt < nk; ++kt) {
    GemmSmem& cur = st[kt & 1];
    wait_async0();       // my async loads into cur are done
    __syncthreads();     // everyone's loads done; everyone done reading other buffer
    if (kt + 1 < nk)     // overlap next tile's global->LDS with this tile's WMMAs
      stage_all_async(Ahi, Alo, Bthi, Btlo, st[(kt + 1) & 1], K,
                      blockRow, blockCol, (kt + 1) * BK, tid);

    bf16x16 ah[4], al[4], bh[2], bl[2];
#pragma unroll
    for (int mi = 0; mi < 4; ++mi) {
      int r = waveM * 64 + mi * 16 + (lane & 15);
      ah[mi] = load_fragA(&cur.ahi[r][0], lane);
      al[mi] = load_fragA(&cur.alo[r][0], lane);
    }
#pragma unroll
    for (int ni = 0; ni < 2; ++ni) {
      int r = waveN * 32 + ni * 16 + (lane & 15);
      bh[ni] = load_fragB(&cur.bhi[r][0], lane);
      bl[ni] = load_fragB(&cur.blo[r][0], lane);
    }
#pragma unroll
    for (int mi = 0; mi < 4; ++mi) {
#pragma unroll
      for (int ni = 0; ni < 2; ++ni) {
        acc[mi][ni] = __builtin_amdgcn_wmma_f32_16x16x32_bf16(
            false, ah[mi], false, bh[ni], (short)0, acc[mi][ni], false, false);
        acc[mi][ni] = __builtin_amdgcn_wmma_f32_16x16x32_bf16(
            false, al[mi], false, bh[ni], (short)0, acc[mi][ni], false, false);
        acc[mi][ni] = __builtin_amdgcn_wmma_f32_16x16x32_bf16(
            false, ah[mi], false, bl[ni], (short)0, acc[mi][ni], false, false);
      }
    }
  }
}

// ---------------- kernels ----------------

__global__ void split_f32_kernel(const float* __restrict__ in,
                                 bf16* __restrict__ hi, bf16* __restrict__ lo, long n) {
  long i = (long)blockIdx.x * blockDim.x + threadIdx.x;
  if (i < n) {
    float v = in[i];
    bf16 h = (bf16)v;
    hi[i] = h;
    lo[i] = (bf16)(v - (float)h);
  }
}

// W[K][N] f32  ->  Wt_hi/Wt_lo[N][K] bf16 (LDS-tiled 32x32 transpose)
__global__ void transpose_split_kernel(const float* __restrict__ W,
                                       bf16* __restrict__ thi, bf16* __restrict__ tlo,
                                       int K, int N) {
  __shared__ float tile[32][33];
  const int n0 = blockIdx.x * 32, k0 = blockIdx.y * 32;
  const int tx = threadIdx.x, ty = threadIdx.y;  // 32 x 8
#pragma unroll
  for (int i = 0; i < 4; ++i) {
    int r = ty + 8 * i;
    tile[r][tx] = W[(long)(k0 + r) * N + (n0 + tx)];
  }
  __syncthreads();
#pragma unroll
  for (int i = 0; i < 4; ++i) {
    int n = n0 + ty + 8 * i;
    float v = tile[tx][ty + 8 * i];
    bf16 h = (bf16)v;
    long o = (long)n * K + (k0 + tx);
    thi[o] = h;
    tlo[o] = (bf16)(v - (float)h);
  }
}

template <bool RELU, bool WF32, bool WSPLIT>
__global__ __launch_bounds__(256) void gemm_bf16x3_kernel(
    const bf16* __restrict__ Ahi, const bf16* __restrict__ Alo,
    const bf16* __restrict__ Bthi, const bf16* __restrict__ Btlo,
    const float* __restrict__ bias,
    float* __restrict__ Cf, bf16* __restrict__ Chi, bf16* __restrict__ Clo,
    int M, int N, int K) {
  __shared__ GemmSmem st[2];               // 81920 B, double buffered
  const int tid = threadIdx.x;
  const long blockRow = (long)blockIdx.y * BM;
  const long blockCol = (long)blockIdx.x * BN;
  floatx8 acc[4][2];
#pragma unroll
  for (int mi = 0; mi < 4; ++mi)
#pragma unroll
    for (int ni = 0; ni < 2; ++ni) acc[mi][ni] = {};

  gemm_mainloop(Ahi, Alo, Bthi, Btlo, st, acc, tid, K, blockRow, blockCol);

  const int lane  = tid & 31;
  const int waveM = (tid >> 5) >> 2;
  const int waveN = (tid >> 5) & 3;
  const int rbase = waveM * 64 + 8 * (lane >> 4);
  const int cbase = lane & 15;
#pragma unroll
  for (int mi = 0; mi < 4; ++mi) {
#pragma unroll
    for (int ni = 0; ni < 2; ++ni) {
      long col = blockCol + waveN * 32 + ni * 16 + cbase;
      float bv = bias ? bias[col] : 0.0f;
#pragma unroll
      for (int j = 0; j < 8; ++j) {
        long row = blockRow + rbase + mi * 16 + j;
        float v = acc[mi][ni][j] + bv;
        if (RELU) v = fmaxf(v, 0.0f);
        long o = row * (long)N + col;
        if (WF32) Cf[o] = v;
        if (WSPLIT) {
          bf16 h = (bf16)v;
          Chi[o] = h;
          Clo[o] = (bf16)(v - (float)h);
        }
      }
    }
  }
}

union VqSmem {
  GemmSmem st[2];             // 81920 B (stage buffers, retired before epilogue)
  float dt[BM][BN];           // 65536 B distance tile
};

// S = z_e @ codebook^T; d = ||c||^2 - 2*S (||z||^2 is row-constant -> argmin invariant)
__global__ __launch_bounds__(256) void vq_gemm_argmin_kernel(
    const bf16* __restrict__ Ahi, const bf16* __restrict__ Alo,
    const bf16* __restrict__ CBhi, const bf16* __restrict__ CBlo,
    const float* __restrict__ cnorm,
    unsigned long long* __restrict__ packed, int M, int N, int K) {
  __shared__ VqSmem sm;
  const int tid = threadIdx.x;
  const long blockRow = (long)blockIdx.y * BM;
  const long blockCol = (long)blockIdx.x * BN;
  floatx8 acc[4][2];
#pragma unroll
  for (int mi = 0; mi < 4; ++mi)
#pragma unroll
    for (int ni = 0; ni < 2; ++ni) acc[mi][ni] = {};

  gemm_mainloop(Ahi, Alo, CBhi, CBlo, sm.st, acc, tid, K, blockRow, blockCol);

  __syncthreads();  // retire staging LDS before aliasing as dt
  const int lane  = tid & 31;
  const int waveM = (tid >> 5) >> 2;
  const int waveN = (tid >> 5) & 3;
  const int rbase = waveM * 64 + 8 * (lane >> 4);
  const int cbase = lane & 15;
#pragma unroll
  for (int mi = 0; mi < 4; ++mi) {
#pragma unroll
    for (int ni = 0; ni < 2; ++ni) {
      int cl = waveN * 32 + ni * 16 + cbase;
      float cn = cnorm[blockCol + cl];
#pragma unroll
      for (int j = 0; j < 8; ++j) {
        int rl = rbase + mi * 16 + j;
        sm.dt[rl][cl] = cn - 2.0f * acc[mi][ni][j];
      }
    }
  }
  __syncthreads();
  if (tid < BM) {
    float best = 3.4e38f;
    int bi = 0;
    for (int i = 0; i < BN; ++i) {          // staggered scan: bank-conflict-free
      int c = (tid + i) & (BN - 1);
      float v = sm.dt[tid][c];
      if (v < best || (v == best && c < bi)) { best = v; bi = c; }
    }
    unsigned long long pk =
        ((unsigned long long)fkey(best) << 32) | (unsigned)(blockCol + bi);
    atomicMin(&packed[blockRow + tid], pk);
  }
}

__global__ void rownorm_kernel(const float* __restrict__ cb, float* __restrict__ cn) {
  int r = blockIdx.x * blockDim.x + threadIdx.x;
  if (r < KCB) {
    float s = 0.0f;
    for (int d = 0; d < DC; ++d) {
      float v = cb[(long)r * DC + d];
      s += v * v;
    }
    cn[r] = s;
  }
}

__global__ void init_packed_kernel(unsigned long long* p, int n) {
  int i = blockIdx.x * blockDim.x + threadIdx.x;
  if (i < n) p[i] = ~0ull;
}

__global__ void vq_gather_kernel(const unsigned long long* __restrict__ packed,
                                 const float* __restrict__ codebook,
                                 float* __restrict__ out_codes,
                                 float* __restrict__ out_idx,
                                 bf16* __restrict__ act_hi, bf16* __restrict__ act_lo) {
  long t = (long)blockIdx.x * blockDim.x + threadIdx.x;
  long b = t >> 7;
  int  d = (int)(t & (DC - 1));
  if (b >= TB) return;
  unsigned idx = (unsigned)(packed[b] & 0xFFFFFFFFull);
  float v = codebook[(long)idx * DC + d];
  long o = b * DC + d;
  out_codes[o] = v;                // selected_codes; also z_q forward value
  bf16 h = (bf16)v;
  act_hi[o] = h;
  act_lo[o] = (bf16)(v - (float)h);
  if (d == 0) out_idx[b] = (float)idx;
}

// ---------------- host ----------------

extern "C" void kernel_launch(void* const* d_in, const int* in_sizes, int n_in,
                              void* d_out, int out_size, void* d_ws, size_t ws_size,
                              hipStream_t stream) {
  (void)in_sizes; (void)n_in; (void)out_size; (void)ws_size;
  const float* x        = (const float*)d_in[0];
  const float* enc_w[4] = {(const float*)d_in[1], (const float*)d_in[3],
                           (const float*)d_in[5], (const float*)d_in[7]};
  const float* enc_b[4] = {(const float*)d_in[2], (const float*)d_in[4],
                           (const float*)d_in[6], (const float*)d_in[8]};
  const float* dec_w[4] = {(const float*)d_in[9],  (const float*)d_in[11],
                           (const float*)d_in[13], (const float*)d_in[15]};
  const float* dec_b[4] = {(const float*)d_in[10], (const float*)d_in[12],
                           (const float*)d_in[14], (const float*)d_in[16]};
  const float* codebook = (const float*)d_in[17];
  float* out = (float*)d_out;

  const long M = TB;
  const size_t OFF_XREC  = 0;
  const size_t OFF_ZE    = (size_t)TB * DIN;
  const size_t OFF_CODES = OFF_ZE + (size_t)TB * DC;
  const size_t OFF_IDX   = OFF_CODES + (size_t)TB * DC;

  uint8_t* base = (uint8_t*)d_ws;
  size_t off = 0;
  auto take = [&](size_t bytes) -> uint8_t* {
    uint8_t* r = base + off;
    off += (bytes + 255) & ~(size_t)255;
    return r;
  };
  const size_t actB = (size_t)TB * HH * sizeof(bf16);
  bf16* A_hi = (bf16*)take(actB);
  bf16* A_lo = (bf16*)take(actB);
  bf16* B_hi = (bf16*)take(actB);
  bf16* B_lo = (bf16*)take(actB);
  bf16* wt_hi = (bf16*)take((size_t)HH * HH * sizeof(bf16));
  bf16* wt_lo = (bf16*)take((size_t)HH * HH * sizeof(bf16));
  bf16* cb_hi = (bf16*)take((size_t)KCB * DC * sizeof(bf16));
  bf16* cb_lo = (bf16*)take((size_t)KCB * DC * sizeof(bf16));
  float* cnorm = (float*)take((size_t)KCB * sizeof(float));
  unsigned long long* packed = (unsigned long long*)take((size_t)TB * 8);

  // split x -> bf16 hi/lo
  {
    long n = M * DIN;
    split_f32_kernel<<<dim3((unsigned)((n + 255) / 256)), dim3(256), 0, stream>>>(x, A_hi, A_lo, n);
  }

  bf16 *cur_hi = A_hi, *cur_lo = A_lo, *nxt_hi = B_hi, *nxt_lo = B_lo;
  auto swap_bufs = [&]() {
    bf16* t;
    t = cur_hi; cur_hi = nxt_hi; nxt_hi = t;
    t = cur_lo; cur_lo = nxt_lo; nxt_lo = t;
  };

  // encoder
  const int encK[4] = {DIN, HH, HH, HH}, encN[4] = {HH, HH, HH, DC};
  for (int i = 0; i < 4; ++i) {
    transpose_split_kernel<<<dim3(encN[i] / 32, encK[i] / 32), dim3(32, 8), 0, stream>>>(
        enc_w[i], wt_hi, wt_lo, encK[i], encN[i]);
    dim3 grid(encN[i] / BN, (unsigned)(M / BM));
    if (i < 3)
      gemm_bf16x3_kernel<true, false, true><<<grid, dim3(256), 0, stream>>>(
          cur_hi, cur_lo, wt_hi, wt_lo, enc_b[i], nullptr, nxt_hi, nxt_lo,
          (int)M, encN[i], encK[i]);
    else  // z_e: f32 to d_out + hi/lo split for VQ
      gemm_bf16x3_kernel<false, true, true><<<grid, dim3(256), 0, stream>>>(
          cur_hi, cur_lo, wt_hi, wt_lo, enc_b[i], out + OFF_ZE, nxt_hi, nxt_lo,
          (int)M, encN[i], encK[i]);
    swap_bufs();
  }

  // vector quantization
  {
    long n = (long)KCB * DC;
    split_f32_kernel<<<dim3((unsigned)((n + 255) / 256)), dim3(256), 0, stream>>>(codebook, cb_hi, cb_lo, n);
  }
  rownorm_kernel<<<dim3((KCB + 127) / 128), dim3(128), 0, stream>>>(codebook, cnorm);
  init_packed_kernel<<<dim3((TB + 255) / 256), dim3(256), 0, stream>>>(packed, TB);
  vq_gemm_argmin_kernel<<<dim3(KCB / BN, (unsigned)(M / BM)), dim3(256), 0, stream>>>(
      cur_hi, cur_lo, cb_hi, cb_lo, cnorm, packed, (int)M, KCB, DC);
  vq_gather_kernel<<<dim3((unsigned)(((long)TB * DC + 255) / 256)), dim3(256), 0, stream>>>(
      packed, codebook, out + OFF_CODES, out + OFF_IDX, nxt_hi, nxt_lo);
  swap_bufs();  // decoder input = codes (straight-through forward)

  // decoder
  const int decK[4] = {DC, HH, HH, HH}, decN[4] = {HH, HH, HH, DIN};
  for (int i = 0; i < 4; ++i) {
    transpose_split_kernel<<<dim3(decN[i] / 32, decK[i] / 32), dim3(32, 8), 0, stream>>>(
        dec_w[i], wt_hi, wt_lo, decK[i], decN[i]);
    dim3 grid(decN[i] / BN, (unsigned)(M / BM));
    if (i < 3)
      gemm_bf16x3_kernel<true, false, true><<<grid, dim3(256), 0, stream>>>(
          cur_hi, cur_lo, wt_hi, wt_lo, dec_b[i], nullptr, nxt_hi, nxt_lo,
          (int)M, decN[i], decK[i]);
    else
      gemm_bf16x3_kernel<false, true, false><<<grid, dim3(256), 0, stream>>>(
          cur_hi, cur_lo, wt_hi, wt_lo, dec_b[i], out + OFF_XREC, nullptr, nullptr,
          (int)M, decN[i], decK[i]);
    swap_bufs();
  }
}
